// GEMS18d_23270132809884
// MI455X (gfx1250) — compile-verified, matrix operand order checked by reading, not run on hardware
//
#include <hip/hip_runtime.h>

typedef __bf16 bf16;
typedef __attribute__((ext_vector_type(16))) __bf16 v16bf;
typedef __attribute__((ext_vector_type(8)))  __bf16 v8bf;
typedef __attribute__((ext_vector_type(8)))  float  v8f;

#define WMMA_BF16(a, b, c) \
  __builtin_amdgcn_wmma_f32_16x16x32_bf16(false, (a), false, (b), (short)0, (c), false, false)

static constexpr int NNODES = 100000;
static constexpr int NEDGES = 800000;
static constexpr int NGRAPH = 256;
static constexpr int INCH   = 128;
static constexpr int HID    = 64;
static constexpr int GLOB   = 384;
static constexpr float EPSV = 1e-5f;

__device__ __forceinline__ void lds_fence() {
  asm volatile("s_wait_dscnt 0" ::: "memory");
}

// ---------------- fragment builders (CDNA5 ISA 7.12.2 layouts) ----------------
// A (16x32 bf16): lane m = lane&15, kh = lane>>4; elem i -> k = k0 + kh*8 + ((i>>3)<<4) + (i&7)

// fast path: full 32-wide chunk, rows 16B-aligned
__device__ __forceinline__ v16bf a_frag_f32v(const float* __restrict__ rowp, int k0, int kh) {
  const float* p = rowp + k0 + kh * 8;
  float4 u0 = *(const float4*)(p);
  float4 u1 = *(const float4*)(p + 4);
  float4 u2 = *(const float4*)(p + 16);
  float4 u3 = *(const float4*)(p + 20);
  v16bf a;
  a[0] = (bf16)u0.x;  a[1] = (bf16)u0.y;  a[2] = (bf16)u0.z;  a[3] = (bf16)u0.w;
  a[4] = (bf16)u1.x;  a[5] = (bf16)u1.y;  a[6] = (bf16)u1.z;  a[7] = (bf16)u1.w;
  a[8] = (bf16)u2.x;  a[9] = (bf16)u2.y;  a[10] = (bf16)u2.z; a[11] = (bf16)u2.w;
  a[12] = (bf16)u3.x; a[13] = (bf16)u3.y; a[14] = (bf16)u3.z; a[15] = (bf16)u3.w;
  return a;
}

// predicated path (ragged K tail)
__device__ __forceinline__ v16bf a_frag_f32(const float* __restrict__ rowp, int k0, int kh, int klim) {
  v16bf a;
#pragma unroll
  for (int i = 0; i < 16; ++i) {
    int k = k0 + kh * 8 + ((i >> 3) << 4) + (i & 7);
    float f = (k < klim) ? rowp[k] : 0.0f;
    a[i] = (bf16)f;
  }
  return a;
}

__device__ __forceinline__ v16bf a_frag_lds(const bf16* __restrict__ rowp, int k0, int kh) {
  const bf16* p = rowp + k0 + kh * 8;
  v8bf lo = *(const v8bf*)(p);
  v8bf hi = *(const v8bf*)(p + 16);
  v16bf a;
#pragma unroll
  for (int i = 0; i < 8; ++i) { a[i] = lo[i]; a[i + 8] = hi[i]; }
  return a;
}

// B fragments come pre-packed: P[((kc*ntn + nt)*32 + lane)*16 .. +15]
__device__ __forceinline__ v16bf b_frag_packed(const bf16* __restrict__ P, int kc, int nt,
                                               int ntn, int lane) {
  return *(const v16bf*)(P + ((size_t)(kc * ntn + nt) * 32 + lane) * 16);
}

__device__ __forceinline__ unsigned fenc(float f) {
  unsigned u = __float_as_uint(f);
  return (u & 0x80000000u) ? ~u : (u | 0x80000000u);
}
__device__ __forceinline__ float fdec(unsigned u) {
  unsigned b = (u & 0x80000000u) ? (u & 0x7fffffffu) : ~u;
  return __uint_as_float(b);
}

// -------- one-time weight repack: W[K,N] f32 -> bf16 WMMA-B fragment layout --------
__global__ void pack_w_kernel(const float* __restrict__ W, bf16* __restrict__ P, int K, int N) {
  int ntn = N >> 4;
  int nkc = (K + 31) >> 5;
  int t = blockIdx.x * blockDim.x + threadIdx.x;
  if (t >= nkc * ntn * 32) return;
  int lane = t & 31;
  int rem = t >> 5;
  int nt = rem % ntn;
  int kc = rem / ntn;
  int n = nt * 16 + (lane & 15);
  int kh = lane >> 4;
  bf16* dst = P + (size_t)t * 16;
#pragma unroll
  for (int i = 0; i < 16; ++i) {
    int k = kc * 32 + kh * 8 + ((i >> 3) << 4) + (i & 7);
    dst[i] = (bf16)((k < K) ? W[(size_t)k * N + n] : 0.0f);
  }
}

// ---------------- node encoder: out = relu(x@W1+b1)@W2+b2 ; x[M,128] ----------------
__global__ void node_mlp_kernel(const float* __restrict__ x,
                                const bf16* __restrict__ pw1, const float* __restrict__ b1,
                                const bf16* __restrict__ pw2, const float* __restrict__ b2,
                                float* __restrict__ out, int M) {
  __shared__ bf16 smem[4][16 * 32];
  int wave = threadIdx.x >> 5, lane = threadIdx.x & 31;
  int n15 = lane & 15, kh = lane >> 4;
  int tile = blockIdx.x * 4 + wave;
  int base = tile * 16;
  const float* arow = x + (long)min(base + n15, M - 1) * INCH;

  v16bf ax[4];
#pragma unroll
  for (int kq = 0; kq < 4; ++kq) ax[kq] = a_frag_f32v(arow, kq * 32, kh);

  v8f acc[4];
#pragma unroll
  for (int t = 0; t < 4; ++t) acc[t] = (v8f){0, 0, 0, 0, 0, 0, 0, 0};

  bf16* sm = &smem[wave][0];
  for (int c2 = 0; c2 < 8; ++c2) {               // 256 hidden in chunks of 32
    v8f h0 = (v8f){0, 0, 0, 0, 0, 0, 0, 0};
    v8f h1 = (v8f){0, 0, 0, 0, 0, 0, 0, 0};
#pragma unroll
    for (int kq = 0; kq < 4; ++kq) {             // pw1: K=128 (4 chunks), N=256 (16 tiles)
      v16bf bA = b_frag_packed(pw1, kq, c2 * 2, 16, lane);
      v16bf bB = b_frag_packed(pw1, kq, c2 * 2 + 1, 16, lane);
      h0 = WMMA_BF16(ax[kq], bA, h0);
      h1 = WMMA_BF16(ax[kq], bB, h1);
    }
    float bb0 = b1[c2 * 32 + n15], bb1 = b1[c2 * 32 + 16 + n15];
#pragma unroll
    for (int j = 0; j < 8; ++j) {
      int mr = j + 8 * kh;
      sm[mr * 32 + n15]      = (bf16)fmaxf(h0[j] + bb0, 0.0f);
      sm[mr * 32 + 16 + n15] = (bf16)fmaxf(h1[j] + bb1, 0.0f);
    }
    lds_fence();
    v16bf a2 = a_frag_lds(sm + n15 * 32, 0, kh);
#pragma unroll
    for (int t = 0; t < 4; ++t) {                // pw2: K=256 (8 chunks), N=64 (4 tiles)
      v16bf b = b_frag_packed(pw2, c2, t, 4, lane);
      acc[t] = WMMA_BF16(a2, b, acc[t]);
    }
    lds_fence();
  }
#pragma unroll
  for (int t = 0; t < 4; ++t) {
    float bb = b2[t * 16 + n15];
#pragma unroll
    for (int j = 0; j < 8; ++j) {
      int r = base + j + 8 * kh;
      if (r < M) out[(long)r * HID + t * 16 + n15] = acc[t][j] + bb;
    }
  }
}

// ---- fused edge MLP: ea_out = relu(cat(xn[row],xn[col],ea)@W1+b1)@W2+b2 ----
__global__ void edge_mlp_kernel(const float* __restrict__ xn, const float* __restrict__ ea_in,
                                const int* __restrict__ row, const int* __restrict__ col,
                                const bf16* __restrict__ pw1, const float* __restrict__ b1,
                                const bf16* __restrict__ pw2, const float* __restrict__ b2,
                                float* __restrict__ ea_out, int E, int ED) {
  __shared__ bf16 smem[4][16 * 64];
  int wave = threadIdx.x >> 5, lane = threadIdx.x & 31;
  int n15 = lane & 15, kh = lane >> 4;
  int tile = blockIdx.x * 4 + wave;
  int base = tile * 16;
  int e = min(base + n15, E - 1);
  const float* prow = xn + (long)row[e] * HID;
  const float* pcol = xn + (long)col[e] * HID;
  const float* pea  = ea_in + (long)e * ED;
  int K1 = 128 + ED;
  int nchunks = (K1 + 31) >> 5;

  v8f h[4];
#pragma unroll
  for (int t = 0; t < 4; ++t) h[t] = (v8f){0, 0, 0, 0, 0, 0, 0, 0};

  for (int kc = 0; kc < nchunks; ++kc) {
    int k0 = kc * 32;
    const float* p; int lk0, klim;
    if (k0 < 64)       { p = prow; lk0 = k0;       klim = 64; }
    else if (k0 < 128) { p = pcol; lk0 = k0 - 64;  klim = 64; }
    else               { p = pea;  lk0 = k0 - 128; klim = ED; }
    v16bf a = (lk0 + 32 <= klim) ? a_frag_f32v(p, lk0, kh)
                                 : a_frag_f32(p, lk0, kh, klim);
#pragma unroll
    for (int t = 0; t < 4; ++t) {
      v16bf b = b_frag_packed(pw1, kc, t, 4, lane);
      h[t] = WMMA_BF16(a, b, h[t]);
    }
  }
  bf16* sm = &smem[wave][0];
#pragma unroll
  for (int t = 0; t < 4; ++t) {
    float bb = b1[t * 16 + n15];
#pragma unroll
    for (int j = 0; j < 8; ++j) {
      int mr = j + 8 * kh;
      sm[mr * 64 + t * 16 + n15] = (bf16)fmaxf(h[t][j] + bb, 0.0f);
    }
  }
  lds_fence();
  v8f o[4];
#pragma unroll
  for (int t = 0; t < 4; ++t) o[t] = (v8f){0, 0, 0, 0, 0, 0, 0, 0};
#pragma unroll
  for (int kc = 0; kc < 2; ++kc) {
    v16bf a2 = a_frag_lds(sm + n15 * 64, kc * 32, kh);
#pragma unroll
    for (int t = 0; t < 4; ++t) {
      v16bf b = b_frag_packed(pw2, kc, t, 4, lane);
      o[t] = WMMA_BF16(a2, b, o[t]);
    }
  }
#pragma unroll
  for (int t = 0; t < 4; ++t) {
    float bb = b2[t * 16 + n15];
#pragma unroll
    for (int j = 0; j < 8; ++j) {
      int r = base + j + 8 * kh;
      if (r < E) ea_out[(long)r * HID + t * 16 + n15] = o[t][j] + bb;
    }
  }
}

// ---------------- out[M,64] = in[M,64] @ W[64,64] + bias ----------------
__global__ void gemm64_kernel(const float* __restrict__ in, const bf16* __restrict__ pW,
                              const float* __restrict__ bias, float* __restrict__ out, int M) {
  int wave = threadIdx.x >> 5, lane = threadIdx.x & 31;
  int n15 = lane & 15, kh = lane >> 4;
  int tile = blockIdx.x * 4 + wave;
  int base = tile * 16;
  const float* arow = in + (long)min(base + n15, M - 1) * HID;
  v8f acc[4];
#pragma unroll
  for (int t = 0; t < 4; ++t) acc[t] = (v8f){0, 0, 0, 0, 0, 0, 0, 0};
#pragma unroll
  for (int kc = 0; kc < 2; ++kc) {
    v16bf a = a_frag_f32v(arow, kc * 32, kh);
#pragma unroll
    for (int t = 0; t < 4; ++t) {
      v16bf b = b_frag_packed(pW, kc, t, 4, lane);
      acc[t] = WMMA_BF16(a, b, acc[t]);
    }
  }
#pragma unroll
  for (int t = 0; t < 4; ++t) {
    float bb = bias[t * 16 + n15];
#pragma unroll
    for (int j = 0; j < 8; ++j) {
      int r = base + j + 8 * kh;
      if (r < M) out[(long)r * HID + t * 16 + n15] = acc[t][j] + bb;
    }
  }
}

// ---------------- segment helpers ----------------
__global__ void deg_kernel(const int* __restrict__ col, float* __restrict__ deg, int E) {
  int e = blockIdx.x * blockDim.x + threadIdx.x;
  if (e < E) atomicAdd(&deg[col[e]], 1.0f);
}

__global__ void loopea_acc_kernel(const float* __restrict__ ea, const int* __restrict__ col,
                                  float* __restrict__ loopea, int E) {
  long t = (long)blockIdx.x * blockDim.x + threadIdx.x;
  if (t >= (long)E * 16) return;
  int e = (int)(t >> 4), d0 = ((int)t & 15) * 4;
  long c = col[e];
#pragma unroll
  for (int i = 0; i < 4; ++i)
    atomicAdd(&loopea[c * HID + d0 + i], ea[(long)e * HID + d0 + i]);
}

__global__ void loopea_div_kernel(float* __restrict__ loopea, const float* __restrict__ deg, int N) {
  long t = (long)blockIdx.x * blockDim.x + threadIdx.x;
  if (t >= (long)N * HID) return;
  loopea[t] /= fmaxf(deg[t >> 6], 1.0f);
}

// ---- GATv2 attention logits: ee = ea2@g_we (WMMA); alpha = dot(lrelu(xl[r]+xr[c]+ee), att)
__global__ void att_alpha_kernel(const float* __restrict__ ea, const float* __restrict__ loopea,
                                 const float* __restrict__ xl, const float* __restrict__ xr,
                                 const int* __restrict__ row, const int* __restrict__ col,
                                 const bf16* __restrict__ pwe, const float* __restrict__ g_att,
                                 float* __restrict__ alpha, unsigned* __restrict__ amax,
                                 int E, int N) {
  int EP = E + N;
  int wave = threadIdx.x >> 5, lane = threadIdx.x & 31;
  int n15 = lane & 15, kh = lane >> 4;
  int tile = blockIdx.x * 4 + wave;
  int base = tile * 16;
  int ep = min(base + n15, EP - 1);
  const float* pea = (ep < E) ? (ea + (long)ep * HID) : (loopea + (long)(ep - E) * HID);

  v8f ee[4];
#pragma unroll
  for (int h = 0; h < 4; ++h) ee[h] = (v8f){0, 0, 0, 0, 0, 0, 0, 0};
#pragma unroll
  for (int kc = 0; kc < 2; ++kc) {
    v16bf a = a_frag_f32v(pea, kc * 32, kh);
#pragma unroll
    for (int h = 0; h < 4; ++h) {
      v16bf b = b_frag_packed(pwe, kc, h, 4, lane);
      ee[h] = WMMA_BF16(a, b, ee[h]);
    }
  }
  float attv[4];
#pragma unroll
  for (int h = 0; h < 4; ++h) attv[h] = g_att[h * 16 + n15];

#pragma unroll
  for (int j = 0; j < 8; ++j) {
    int mr = j + 8 * kh;
    int epj = base + mr;
    bool valid = epj < EP;
    int epc = min(epj, EP - 1);
    int r, c;
    if (epc < E) { r = row[epc]; c = col[epc]; } else { r = c = epc - E; }
#pragma unroll
    for (int h = 0; h < 4; ++h) {
      float mv = xl[(long)r * HID + h * 16 + n15] + xr[(long)c * HID + h * 16 + n15] + ee[h][j];
      mv = (mv > 0.0f) ? mv : 0.2f * mv;     // leaky_relu 0.2
      float p = mv * attv[h];
      p += __shfl_xor(p, 1, 32);
      p += __shfl_xor(p, 2, 32);
      p += __shfl_xor(p, 4, 32);
      p += __shfl_xor(p, 8, 32);
      if (n15 == 0 && valid) {
        alpha[(long)epj * 4 + h] = p;
        atomicMax(&amax[(long)c * 4 + h], fenc(p));
      }
    }
  }
}

__global__ void att_norm_kernel(float* __restrict__ alpha, const int* __restrict__ col,
                                const unsigned* __restrict__ amax, float* __restrict__ den,
                                int E, int N) {
  long t = (long)blockIdx.x * blockDim.x + threadIdx.x;
  int EP = E + N;
  if (t >= (long)EP * 4) return;
  int ep = (int)(t >> 2), h = (int)t & 3;
  int c = (ep < E) ? col[ep] : (ep - E);
  float ex = __expf(alpha[t] - fdec(amax[(long)c * 4 + h]));
  alpha[t] = ex;
  atomicAdd(&den[(long)c * 4 + h], ex);
}

__global__ void att_agg_kernel(const float* __restrict__ alpha, const float* __restrict__ den,
                               const float* __restrict__ xl, const int* __restrict__ row,
                               const int* __restrict__ col, float* __restrict__ acc,
                               int E, int N) {
  long t = (long)blockIdx.x * blockDim.x + threadIdx.x;
  int EP = E + N;
  if (t >= (long)EP * HID) return;
  int ep = (int)(t >> 6), d = (int)t & 63, h = d >> 4;
  int r, c;
  if (ep < E) { r = row[ep]; c = col[ep]; } else { r = c = ep - E; }
  float w = alpha[(long)ep * 4 + h] / den[(long)c * 4 + h];
  atomicAdd(&acc[(long)c * HID + d], xl[(long)r * HID + d] * w);
}

__global__ void node_fin_kernel(float* __restrict__ acc, const float* __restrict__ gbias, int N) {
  long t = (long)blockIdx.x * blockDim.x + threadIdx.x;
  if (t >= (long)N * HID) return;
  acc[t] = fmaxf(acc[t] + gbias[(int)t & 63], 0.0f);
}

__global__ void pool_kernel(const float* __restrict__ x, const int* __restrict__ batch,
                            float* __restrict__ pooled, int N) {
  long t = (long)blockIdx.x * blockDim.x + threadIdx.x;
  if (t >= (long)N * HID) return;
  atomicAdd(&pooled[(long)batch[t >> 6] * HID + ((int)t & 63)], x[t]);
}

// ---------------- global model: u = relu(cat(u,pool)@W1+b1)@W2+b2 ----------------
__global__ void u_mlp_kernel(const float* __restrict__ u_in, const float* __restrict__ pooled,
                             const float* __restrict__ w1, const float* __restrict__ b1,
                             const float* __restrict__ w2, const float* __restrict__ b2,
                             float* __restrict__ u_out) {
  __shared__ float hsh[GLOB];
  int g = blockIdx.x, t = threadIdx.x;
  float s = b1[t];
  for (int k = 0; k < GLOB; ++k) s += u_in[(long)g * GLOB + k] * w1[k * GLOB + t];
  for (int k = 0; k < HID; ++k)  s += pooled[(long)g * HID + k] * w1[(GLOB + k) * GLOB + t];
  hsh[t] = fmaxf(s, 0.0f);
  __syncthreads();
  float o = b2[t];
  for (int k = 0; k < GLOB; ++k) o += hsh[k] * w2[k * GLOB + t];
  u_out[(long)g * GLOB + t] = o;
}

__global__ void bn_kernel(float* __restrict__ z, const float* __restrict__ scale,
                          const float* __restrict__ bias, const float* __restrict__ mean,
                          const float* __restrict__ var, long total, int D) {
  long t = (long)blockIdx.x * blockDim.x + threadIdx.x;
  if (t >= total) return;
  int d = (int)(t % D);
  z[t] = (z[t] - mean[d]) * rsqrtf(var[d] + EPSV) * scale[d] + bias[d];
}

__global__ void head_kernel(const float* __restrict__ u, const float* __restrict__ w1,
                            const float* __restrict__ b1, const float* __restrict__ w2,
                            const float* __restrict__ b2, float* __restrict__ out) {
  __shared__ float hsh[64];
  int g = blockIdx.x, t = threadIdx.x;   // 64 threads
  float s = b1[t];
  for (int k = 0; k < GLOB; ++k) s += u[(long)g * GLOB + k] * w1[k * 64 + t];
  hsh[t] = fmaxf(s, 0.0f) * w2[t];       // fold fc2 weight
  __syncthreads();
  if (t == 0) {
    float o = b2[0];
    for (int k = 0; k < 64; ++k) o += hsh[k];
    out[g] = o;
  }
}

// =====================================================================================
extern "C" void kernel_launch(void* const* d_in, const int* in_sizes, int n_in,
                              void* d_out, int out_size, void* d_ws, size_t ws_size,
                              hipStream_t stream) {
  (void)in_sizes; (void)n_in; (void)out_size; (void)ws_size;
  const float* x    = (const float*)d_in[0];
  const int*   ei   = (const int*)d_in[1];
  const int*   row  = ei;
  const int*   col  = ei + NEDGES;
  const float* eatt = (const float*)d_in[2];
  const float* lig  = (const float*)d_in[3];
  const int*   batch = (const int*)d_in[4];

  // params flattened in reference insertion order starting at index 5
  const float* nt_w1 = (const float*)d_in[5];
  const float* nt_b1 = (const float*)d_in[6];
  const float* nt_w2 = (const float*)d_in[7];
  const float* nt_b2 = (const float*)d_in[8];
  auto L = [&](int base, int k) { return (const float*)d_in[base + k]; };
  // layer: e_w1 0, e_b1 1, e_w2 2, e_b2 3, g_wl 4, g_bl 5, g_wr 6, g_br 7,
  //        g_we 8, g_att 9, g_bias 10, u_w1 11, u_b1 12, u_w2 13, u_b2 14
  const int L1 = 9, BNX = 24, BNE = 28, BNU = 32, L2 = 36;
  const float* fc1_w = (const float*)d_in[51];
  const float* fc1_b = (const float*)d_in[52];
  const float* fc2_w = (const float*)d_in[53];
  const float* fc2_b = (const float*)d_in[54];

  // ---------------- workspace carve-up ----------------
  char* w = (char*)d_ws;
  auto take = [&](size_t bytes) { void* p = w; w += (bytes + 255) & ~size_t(255); return p; };
  float*    NODE0  = (float*)take((size_t)NNODES * HID * 4);
  float*    NODE1  = (float*)take((size_t)NNODES * HID * 4);
  float*    XL     = (float*)take((size_t)NNODES * HID * 4);
  float*    XR     = (float*)take((size_t)NNODES * HID * 4);
  float*    LOOPEA = (float*)take((size_t)NNODES * HID * 4);
  float*    EA     = (float*)take((size_t)NEDGES * HID * 4);
  float*    ALPHA  = (float*)take((size_t)(NEDGES + NNODES) * 4 * 4);
  float*    DEG    = (float*)take((size_t)NNODES * 4);
  unsigned* AMAX   = (unsigned*)take((size_t)NNODES * 4 * 4);
  float*    DEN    = (float*)take((size_t)NNODES * 4 * 4);
  float*    POOLED = (float*)take((size_t)NGRAPH * HID * 4);
  float*    U1     = (float*)take((size_t)NGRAPH * GLOB * 4);
  float*    U2     = (float*)take((size_t)NGRAPH * GLOB * 4);

  // packed bf16 weight fragments: nkc(K)*ntn(N)*32 lanes * 16 elems * 2B
  auto psz = [](int K, int N) { return (size_t)(((K + 31) >> 5) * (N >> 4)) * 32 * 16 * 2; };
  bf16* P_ntw1 = (bf16*)take(psz(128, 256));
  bf16* P_ntw2 = (bf16*)take(psz(256, 64));
  bf16* P_ew1[2]  = { (bf16*)take(psz(144, 64)), (bf16*)take(psz(192, 64)) };
  bf16* P_ew2[2]  = { (bf16*)take(psz(64, 64)),  (bf16*)take(psz(64, 64)) };
  bf16* P_gwl[2]  = { (bf16*)take(psz(64, 64)),  (bf16*)take(psz(64, 64)) };
  bf16* P_gwr[2]  = { (bf16*)take(psz(64, 64)),  (bf16*)take(psz(64, 64)) };
  bf16* P_gwe[2]  = { (bf16*)take(psz(64, 64)),  (bf16*)take(psz(64, 64)) };

  dim3 blk128(128), blk256(256);
  auto tb  = [](int tiles) { return dim3((tiles + 3) / 4); };            // 4 waves/block
  auto th  = [](long n)    { return dim3((unsigned)((n + 255) / 256)); };
  auto pgrid = [](int K, int N) { int t = ((K + 31) >> 5) * (N >> 4) * 32; return dim3((t + 255) / 256); };

  // 0) repack all GEMM weights into WMMA-B fragment layout (runs once per launch)
  pack_w_kernel<<<pgrid(128, 256), blk256, 0, stream>>>(nt_w1, P_ntw1, 128, 256);
  pack_w_kernel<<<pgrid(256, 64),  blk256, 0, stream>>>(nt_w2, P_ntw2, 256, 64);
  for (int l = 0; l < 2; ++l) {
    int LB = l ? L2 : L1;
    int K1 = l ? 192 : 144;
    pack_w_kernel<<<pgrid(K1, 64), blk256, 0, stream>>>(L(LB, 0), P_ew1[l], K1, 64);
    pack_w_kernel<<<pgrid(64, 64), blk256, 0, stream>>>(L(LB, 2), P_ew2[l], 64, 64);
    pack_w_kernel<<<pgrid(64, 64), blk256, 0, stream>>>(L(LB, 4), P_gwl[l], 64, 64);
    pack_w_kernel<<<pgrid(64, 64), blk256, 0, stream>>>(L(LB, 6), P_gwr[l], 64, 64);
    pack_w_kernel<<<pgrid(64, 64), blk256, 0, stream>>>(L(LB, 8), P_gwe[l], 64, 64);
  }

  const int EP = NEDGES + NNODES;
  const int nodeTiles = NNODES / 16, edgeTiles = NEDGES / 16, epTiles = EP / 16;

  // 1) node encoder
  node_mlp_kernel<<<tb(nodeTiles), blk128, 0, stream>>>(x, P_ntw1, nt_b1, P_ntw2, nt_b2, NODE0, NNODES);

  for (int layer = 0; layer < 2; ++layer) {
    const int   LB   = layer ? L2 : L1;
    const float* xin = layer ? NODE1 : NODE0;
    float*       acc = layer ? NODE0 : NODE1;
    const float* ein = layer ? EA : eatt;
    const int    ED  = layer ? HID : 16;

    hipMemsetAsync(LOOPEA, 0, (size_t)NNODES * HID * 4, stream);
    hipMemsetAsync(AMAX,   0, (size_t)NNODES * 4 * 4, stream);
    hipMemsetAsync(DEN,    0, (size_t)NNODES * 4 * 4, stream);
    hipMemsetAsync(acc,    0, (size_t)NNODES * HID * 4, stream);
    hipMemsetAsync(POOLED, 0, (size_t)NGRAPH * HID * 4, stream);
    if (layer == 0) hipMemsetAsync(DEG, 0, (size_t)NNODES * 4, stream);

    // edge model (in-place safe: each edge consumes only its own input row)
    edge_mlp_kernel<<<tb(edgeTiles), blk128, 0, stream>>>(
        xin, ein, row, col, P_ew1[layer], L(LB, 1), P_ew2[layer], L(LB, 3), EA, NEDGES, ED);

    if (layer == 0) deg_kernel<<<th(NEDGES), blk256, 0, stream>>>(col, DEG, NEDGES);
    loopea_acc_kernel<<<th((long)NEDGES * 16), blk256, 0, stream>>>(EA, col, LOOPEA, NEDGES);
    loopea_div_kernel<<<th((long)NNODES * HID), blk256, 0, stream>>>(LOOPEA, DEG, NNODES);

    gemm64_kernel<<<tb(nodeTiles), blk128, 0, stream>>>(xin, P_gwl[layer], L(LB, 5), XL, NNODES);
    gemm64_kernel<<<tb(nodeTiles), blk128, 0, stream>>>(xin, P_gwr[layer], L(LB, 7), XR, NNODES);

    att_alpha_kernel<<<tb(epTiles), blk128, 0, stream>>>(
        EA, LOOPEA, XL, XR, row, col, P_gwe[layer], L(LB, 9), ALPHA, AMAX, NEDGES, NNODES);
    att_norm_kernel<<<th((long)EP * 4), blk256, 0, stream>>>(ALPHA, col, AMAX, DEN, NEDGES, NNODES);
    att_agg_kernel<<<th((long)EP * HID), blk256, 0, stream>>>(ALPHA, DEN, XL, row, col, acc, NEDGES, NNODES);
    node_fin_kernel<<<th((long)NNODES * HID), blk256, 0, stream>>>(acc, L(LB, 10), NNODES);

    pool_kernel<<<th((long)NNODES * HID), blk256, 0, stream>>>(acc, batch, POOLED, NNODES);
    u_mlp_kernel<<<dim3(NGRAPH), dim3(GLOB), 0, stream>>>(
        layer ? U1 : lig, POOLED, L(LB, 11), L(LB, 12), L(LB, 13), L(LB, 14), layer ? U2 : U1);

    if (layer == 0) {
      bn_kernel<<<th((long)NNODES * HID), blk256, 0, stream>>>(
          NODE1, (const float*)d_in[BNX], (const float*)d_in[BNX + 1],
          (const float*)d_in[BNX + 2], (const float*)d_in[BNX + 3], (long)NNODES * HID, HID);
      bn_kernel<<<th((long)NEDGES * HID), blk256, 0, stream>>>(
          EA, (const float*)d_in[BNE], (const float*)d_in[BNE + 1],
          (const float*)d_in[BNE + 2], (const float*)d_in[BNE + 3], (long)NEDGES * HID, HID);
      bn_kernel<<<th((long)NGRAPH * GLOB), blk256, 0, stream>>>(
          U1, (const float*)d_in[BNU], (const float*)d_in[BNU + 1],
          (const float*)d_in[BNU + 2], (const float*)d_in[BNU + 3], (long)NGRAPH * GLOB, GLOB);
    }
  }

  head_kernel<<<dim3(NGRAPH), dim3(64), 0, stream>>>(U2, fc1_w, fc1_b, fc2_w, fc2_b, (float*)d_out);
}
